// EnPropPred_2259152797781
// MI455X (gfx1250) — matrix operand order repaired and checked by reading, not compiled, not run on hardware
//
#include <hip/hip_runtime.h>
#include <hip/hip_bf16.h>

typedef _Float16 f16;
typedef __attribute__((ext_vector_type(16))) f16   v16h;
typedef __attribute__((ext_vector_type(8)))  f16   v8h;
typedef __attribute__((ext_vector_type(8)))  float v8f;

#define NN    50000
#define NE    800000
#define NG    2500
#define HID   128
#define NRBF  16
#define MT_N  3125    /* NN/16  */
#define MT_E  50000   /* NE/16  */

// ---------------- WMMA fragment helpers (CDNA5 16x16x32 f16) ----------------
// A (16xK, row-major f16, stride ld): lanes 0-15 -> M=lane, K = k0+{0..7,16..23}
//                                     lanes 16-31 -> M=lane-16, K = k0+{8..15,24..31}
__device__ __forceinline__ v16h frag_a(const f16* base, int ld, int lane, int k0) {
  int m  = lane & 15;
  int kb = k0 + ((lane & 16) ? 8 : 0);
  const f16* p = base + m * ld + kb;
  v8h lo = *(const v8h*)(p);
  v8h hi = *(const v8h*)(p + 16);
  v16h r;
#pragma unroll
  for (int i = 0; i < 8; ++i) { r[i] = lo[i]; r[i + 8] = hi[i]; }
  return r;
}

// Same A layout but sourced from an f32 matrix (converted in-register).
__device__ __forceinline__ v16h frag_a_f32(const float* base, int ld, int lane, int k0) {
  int m  = lane & 15;
  int kb = k0 + ((lane & 16) ? 8 : 0);
  const float* p = base + m * ld + kb;
  v16h r;
#pragma unroll
  for (int i = 0; i < 8; ++i) { r[i] = (f16)p[i]; r[i + 8] = (f16)p[i + 16]; }
  return r;
}

// B (Kx16) from TRANSPOSED weights WT[n][k] (row-major, stride ld), pre-offset
// to the 16-column tile: lanes 0-15 -> N=lane, K=k0+{0..7,16..23} ; lanes 16-31 -> K=k0+{8..15,24..31}
__device__ __forceinline__ v16h frag_b(const f16* wt, int ld, int lane, int k0) {
  int n  = lane & 15;
  int kb = k0 + ((lane & 16) ? 8 : 0);
  const f16* p = wt + n * ld + kb;
  v8h lo = *(const v8h*)(p);
  v8h hi = *(const v8h*)(p + 16);
  v16h r;
#pragma unroll
  for (int i = 0; i < 8; ++i) { r[i] = lo[i]; r[i + 8] = hi[i]; }
  return r;
}

__device__ __forceinline__ v8f wmma32(v16h a, v16h b, v8f c) {
  return __builtin_amdgcn_wmma_f32_16x16x32_f16(false, a, false, b, (short)0, c,
                                                false, false);
}
// C/D layout: lane<16 -> (M=r, N=lane) ; lane>=16 -> (M=r+8, N=lane-16)

// ---------------- node featurization + embedding ----------------
__global__ void k_node_embed(const int* __restrict__ typ,
                             const float* __restrict__ w, const float* __restrict__ b,
                             float* __restrict__ h, f16* __restrict__ h16) {
  int i = blockIdx.x * blockDim.x + threadIdx.x;
  if (i >= NN * HID) return;
  int node = i >> 7, c = i & 127;
  int t = typ[node];
  float charge = (t == 0) ? 1.0f : (float)(t + 5);   // {1,6,7,8,9}
  float q = charge / 9.0f;
  float v = b[c], qp = 1.0f;
#pragma unroll
  for (int p = 0; p < 3; ++p) { v += qp * w[(t * 3 + p) * HID + c]; qp *= q; }
  h[i] = v;
  h16[i] = (f16)v;
}

// ---------------- RBF expansion of edge distances ----------------
__global__ void k_rbf(const float* __restrict__ pos, const int* __restrict__ src,
                      const int* __restrict__ dst, float* __restrict__ rbf) {
  int e = blockIdx.x * blockDim.x + threadIdx.x;
  if (e >= NE) return;
  int s = src[e], d = dst[e];
  float dx = pos[d * 3 + 0] - pos[s * 3 + 0];
  float dy = pos[d * 3 + 1] - pos[s * 3 + 1];
  float dz = pos[d * 3 + 2] - pos[s * 3 + 2];
  float dist = sqrtf(dx * dx + dy * dy + dz * dz + 1e-8f);
  const float coeff = -0.5f / ((10.0f / NRBF) * (10.0f / NRBF));
  const float cstep = 10.0f / (NRBF - 1);
#pragma unroll
  for (int r = 0; r < NRBF; ++r) {
    float t = dist - r * cstep;
    rbf[e * NRBF + r] = __expf(coeff * t * t);
  }
}

// ---------------- per-layer weight prep (f16 transposes + 4-row table) -------
__global__ void k_prep_layer(const float* __restrict__ e1w, const float* __restrict__ e1b,
                             const float* __restrict__ e2w, const float* __restrict__ n1w,
                             const float* __restrict__ n2w, const float* __restrict__ eemb,
                             f16* __restrict__ w1aT, f16* __restrict__ w1bT,
                             float* __restrict__ w1c, float* __restrict__ t4,
                             f16* __restrict__ e2wT, f16* __restrict__ n1wT,
                             f16* __restrict__ n2wT) {
  int tid = blockIdx.x * blockDim.x + threadIdx.x;       // 32768 threads
  if (tid < HID * HID) {
    int n = tid >> 7, k = tid & 127;
    w1aT[n * HID + k] = (f16)e1w[k * HID + n];
    w1bT[n * HID + k] = (f16)e1w[(128 + k) * HID + n];
    e2wT[n * HID + k] = (f16)e2w[k * HID + n];
    n2wT[n * HID + k] = (f16)n2w[k * HID + n];
  }
  if (tid < HID * 256) {
    int n = tid >> 8, k = tid & 255;
    n1wT[n * 256 + k] = (f16)n1w[k * HID + n];
  }
  if (tid < NRBF * HID) w1c[tid] = e1w[(256 + (tid >> 7)) * HID + (tid & 127)];
  if (tid < 4 * HID) {
    int t = tid >> 7, c = tid & 127;
    float v = e1b[c];
    for (int j = 0; j < HID; ++j) v += eemb[t * HID + j] * e1w[(272 + j) * HID + c];
    t4[tid] = v;
  }
}

// ---------------- node projections Pd = h@W1a, Ps = h@W1b (WMMA) ------------
__global__ void __launch_bounds__(256) k_node_proj(const f16* __restrict__ h16,
                                                   const f16* __restrict__ w1aT,
                                                   const f16* __restrict__ w1bT,
                                                   float* __restrict__ Pd,
                                                   float* __restrict__ Ps) {
  int wave = threadIdx.x >> 5, lane = threadIdx.x & 31;
  int mt = blockIdx.x * 8 + wave;
  if (mt >= MT_N) return;
  int row = mt * 16;
  v16h a[4];
#pragma unroll
  for (int s = 0; s < 4; ++s) a[s] = frag_a(h16 + row * HID, HID, lane, s * 32);
  int n0 = lane & 15, mb = (lane & 16) ? 8 : 0;
#pragma unroll
  for (int nt = 0; nt < 8; ++nt) {
    v8f accd = {}, accs = {};
#pragma unroll
    for (int s = 0; s < 4; ++s) {
      accd = wmma32(a[s], frag_b(w1aT + nt * 16 * HID, HID, lane, s * 32), accd);
      accs = wmma32(a[s], frag_b(w1bT + nt * 16 * HID, HID, lane, s * 32), accs);
    }
    int n = nt * 16 + n0;
#pragma unroll
    for (int r = 0; r < 8; ++r) {
      Pd[(row + mb + r) * HID + n] = accd[r];
      Ps[(row + mb + r) * HID + n] = accs[r];
    }
  }
}

// ------- fused edge kernel: build relu(m1) tile in LDS (double-buffered),
// ------- WMMA with register-resident e2 fragments, scatter via f32 atomics
__global__ void __launch_bounds__(256) k_edge(const float* __restrict__ Pd,
                                              const float* __restrict__ Ps,
                                              const float* __restrict__ rbf,
                                              const int* __restrict__ esrc,
                                              const int* __restrict__ edst,
                                              const int* __restrict__ etyp,
                                              const float* __restrict__ w1c,
                                              const float* __restrict__ t4,
                                              const f16* __restrict__ e2wT,
                                              const float* __restrict__ e2b,
                                              float* __restrict__ agg) {
  __shared__ f16   As[2][16 * HID];     // 2 x 4 KB ping-pong tile
  __shared__ float w1cs[NRBF * HID];    // 8 KB
  __shared__ float t4s[4 * HID];        // 2 KB
  int tid = threadIdx.x;
  for (int i = tid; i < NRBF * HID; i += 256) w1cs[i] = w1c[i];
  if (tid < 4 * HID) t4s[tid] = t4[tid];

  int wave = tid >> 5, lane = tid & 31;
  // loop-invariant per wave: B fragments (16 output channels) + bias
  v16h bfrag[4];
#pragma unroll
  for (int s = 0; s < 4; ++s)
    bfrag[s] = frag_b(e2wT + wave * 16 * HID, HID, lane, s * 32);
  int n = wave * 16 + (lane & 15);
  float bias = e2b[n];
  int mb = (lane & 16) ? 8 : 0;
  int c = tid & 127, m0 = tid >> 7;
  __syncthreads();

  int buf = 0;
  for (int tile = blockIdx.x; tile < MT_E; tile += gridDim.x, buf ^= 1) {
    int ebase = tile * 16;
    // prefetch next tile's RBF rows and edge indices (global_prefetch_b8)
    int nbase = ebase + (int)gridDim.x * 16;
    if (nbase < NE && tid < 16) {
      __builtin_prefetch(rbf + (size_t)(nbase + tid) * NRBF, 0, 1);
      __builtin_prefetch(esrc + nbase, 0, 1);
      __builtin_prefetch(edst + nbase, 0, 1);
    }
    // build A tile: m1 = Pd[dst] + Ps[src] + rbf@W1c + T4[type]  -> relu -> f16
#pragma unroll
    for (int j = 0; j < 8; ++j) {
      int m = m0 + j * 2;
      int e = ebase + m;
      int s = esrc[e], d = edst[e], t = etyp[e];
      float v = Pd[d * HID + c] + Ps[s * HID + c] + t4s[t * HID + c];
      const float* rb = rbf + (size_t)e * NRBF;
#pragma unroll
      for (int r = 0; r < NRBF; ++r) v += rb[r] * w1cs[r * HID + c];
      As[buf][m * HID + c] = (f16)fmaxf(v, 0.0f);
    }
    __syncthreads();
    v8f acc = {};
#pragma unroll
    for (int s = 0; s < 4; ++s)
      acc = wmma32(frag_a((const f16*)As[buf], HID, lane, s * 32), bfrag[s], acc);
#pragma unroll
    for (int r = 0; r < 8; ++r) {
      int e = ebase + mb + r;
      atomicAdd(&agg[edst[e] * HID + n], acc[r] + bias);
    }
    // no trailing barrier: next iteration writes the other LDS buffer
  }
}

// ---- node update stage 1: hid = relu([h,agg] @ n1w + n1b)  (K=256) ----
// agg consumed directly as f32 (converted in-register), no cvt pass needed.
__global__ void __launch_bounds__(256) k_node_hidden(const f16* __restrict__ h16,
                                                     const float* __restrict__ agg,
                                                     const f16* __restrict__ n1wT,
                                                     const float* __restrict__ n1b,
                                                     f16* __restrict__ hid16) {
  int wave = threadIdx.x >> 5, lane = threadIdx.x & 31;
  int mt = blockIdx.x * 8 + wave;
  if (mt >= MT_N) return;
  int row = mt * 16;
  v16h ah[4], ag[4];
#pragma unroll
  for (int s = 0; s < 4; ++s) {
    ah[s] = frag_a(h16 + row * HID, HID, lane, s * 32);
    ag[s] = frag_a_f32(agg + row * HID, HID, lane, s * 32);
  }
  int n0 = lane & 15, mb = (lane & 16) ? 8 : 0;
#pragma unroll
  for (int nt = 0; nt < 8; ++nt) {
    const f16* wt = n1wT + nt * 16 * 256;
    v8f acc = {};
#pragma unroll
    for (int s = 0; s < 4; ++s) acc = wmma32(ah[s], frag_b(wt, 256, lane, s * 32), acc);
#pragma unroll
    for (int s = 0; s < 4; ++s) acc = wmma32(ag[s], frag_b(wt, 256, lane, 128 + s * 32), acc);
    int n = nt * 16 + n0;
    float bias = n1b[n];
#pragma unroll
    for (int r = 0; r < 8; ++r)
      hid16[(row + mb + r) * HID + n] = (f16)fmaxf(acc[r] + bias, 0.0f);
  }
}

// ---- node update stage 2: h += hid @ n2w + n2b ; refresh h16 ----
__global__ void __launch_bounds__(256) k_node_out(const f16* __restrict__ hid16,
                                                  const f16* __restrict__ n2wT,
                                                  const float* __restrict__ n2b,
                                                  float* __restrict__ h,
                                                  f16* __restrict__ h16) {
  int wave = threadIdx.x >> 5, lane = threadIdx.x & 31;
  int mt = blockIdx.x * 8 + wave;
  if (mt >= MT_N) return;
  int row = mt * 16;
  v16h a[4];
#pragma unroll
  for (int s = 0; s < 4; ++s) a[s] = frag_a(hid16 + row * HID, HID, lane, s * 32);
  int n0 = lane & 15, mb = (lane & 16) ? 8 : 0;
#pragma unroll
  for (int nt = 0; nt < 8; ++nt) {
    v8f acc = {};
#pragma unroll
    for (int s = 0; s < 4; ++s)
      acc = wmma32(a[s], frag_b(n2wT + nt * 16 * HID, HID, lane, s * 32), acc);
    int n = nt * 16 + n0;
    float bias = n2b[n];
#pragma unroll
    for (int r = 0; r < 8; ++r) {
      int ix = (row + mb + r) * HID + n;
      float v = h[ix] + acc[r] + bias;
      h[ix] = v;
      h16[ix] = (f16)v;
    }
  }
}

// ---- readout stage 1 weights ----
__global__ void k_prep_out(const float* __restrict__ o1w, f16* __restrict__ o1wT) {
  int tid = blockIdx.x * blockDim.x + threadIdx.x;
  if (tid < HID * HID) {
    int n = tid >> 7, k = tid & 127;
    o1wT[n * HID + k] = (f16)o1w[k * HID + n];
  }
}

// ---- readout stage 1: hid = relu(h @ out1w + out1b) ----
__global__ void __launch_bounds__(256) k_out_hidden(const f16* __restrict__ h16,
                                                    const f16* __restrict__ o1wT,
                                                    const float* __restrict__ o1b,
                                                    float* __restrict__ hid) {
  int wave = threadIdx.x >> 5, lane = threadIdx.x & 31;
  int mt = blockIdx.x * 8 + wave;
  if (mt >= MT_N) return;
  int row = mt * 16;
  v16h a[4];
#pragma unroll
  for (int s = 0; s < 4; ++s) a[s] = frag_a(h16 + row * HID, HID, lane, s * 32);
  int n0 = lane & 15, mb = (lane & 16) ? 8 : 0;
#pragma unroll
  for (int nt = 0; nt < 8; ++nt) {
    v8f acc = {};
#pragma unroll
    for (int s = 0; s < 4; ++s)
      acc = wmma32(a[s], frag_b(o1wT + nt * 16 * HID, HID, lane, s * 32), acc);
    int n = nt * 16 + n0;
    float bias = o1b[n];
#pragma unroll
    for (int r = 0; r < 8; ++r)
      hid[(row + mb + r) * HID + n] = fmaxf(acc[r] + bias, 0.0f);
  }
}

// ---- readout stage 2: o = hid @ out2w + out2b ; segment-sum over graphs ----
__global__ void k_readout(const float* __restrict__ hid, const float* __restrict__ o2w,
                          const float* __restrict__ o2b, const int* __restrict__ batch,
                          float* __restrict__ out) {
  int node = blockIdx.x * blockDim.x + threadIdx.x;
  if (node >= NN) return;
  float v = o2b[0];
  for (int c = 0; c < HID; ++c) v += hid[node * HID + c] * o2w[c];
  atomicAdd(&out[batch[node]], v);
}

__global__ void k_copy_pos(const float* __restrict__ pos, float* __restrict__ out) {
  int i = blockIdx.x * blockDim.x + threadIdx.x;
  if (i < NN * 3) out[NG + i] = pos[i];
}

// ---------------------------------------------------------------------------
extern "C" void kernel_launch(void* const* d_in, const int* in_sizes, int n_in,
                              void* d_out, int out_size, void* d_ws, size_t ws_size,
                              hipStream_t stream) {
  const float* pos   = (const float*)d_in[0];
  const int*   typ   = (const int*)d_in[1];
  const int*   esrc  = (const int*)d_in[2];
  const int*   edst  = (const int*)d_in[3];
  const int*   etyp  = (const int*)d_in[4];
  const int*   batch = (const int*)d_in[5];
  // params flattened in jax pytree (sorted-key) order:
  // edge_emb, layers[l]{e1.b,e1.w,e2.b,e2.w,n1.b,n1.w,n2.b,n2.w}, node_emb{b,w},
  // out1{b,w}, out2{b,w}
  const float* eemb = (const float*)d_in[6];
  struct LP { const float *e1b, *e1w, *e2b, *e2w, *n1b, *n1w, *n2b, *n2w; } lp[3];
  int idx = 7;
  for (int l = 0; l < 3; ++l) {
    lp[l].e1b = (const float*)d_in[idx++]; lp[l].e1w = (const float*)d_in[idx++];
    lp[l].e2b = (const float*)d_in[idx++]; lp[l].e2w = (const float*)d_in[idx++];
    lp[l].n1b = (const float*)d_in[idx++]; lp[l].n1w = (const float*)d_in[idx++];
    lp[l].n2b = (const float*)d_in[idx++]; lp[l].n2w = (const float*)d_in[idx++];
  }
  const float* nembb = (const float*)d_in[idx++];
  const float* nembw = (const float*)d_in[idx++];
  const float* o1b   = (const float*)d_in[idx++];
  const float* o1w   = (const float*)d_in[idx++];
  const float* o2b   = (const float*)d_in[idx++];
  const float* o2w   = (const float*)d_in[idx++];

  // ---- carve workspace ----
  char* cur = (char*)d_ws;
  auto carve = [&](size_t bytes) -> void* {
    void* r = (void*)cur;
    cur += (bytes + 255) & ~(size_t)255;
    return r;
  };
  float* h     = (float*)carve((size_t)NN * HID * 4);
  f16*   h16   = (f16*)  carve((size_t)NN * HID * 2);
  float* Pd    = (float*)carve((size_t)NN * HID * 4);
  float* Ps    = (float*)carve((size_t)NN * HID * 4);
  float* agg   = (float*)carve((size_t)NN * HID * 4);
  f16*   hid16 = (f16*)  carve((size_t)NN * HID * 2);
  float* rbf   = (float*)carve((size_t)NE * NRBF * 4);
  f16*   w1aT  = (f16*)  carve(HID * HID * 2);
  f16*   w1bT  = (f16*)  carve(HID * HID * 2);
  float* w1c   = (float*)carve(NRBF * HID * 4);
  float* t4    = (float*)carve(4 * HID * 4);
  f16*   e2wT  = (f16*)  carve(HID * HID * 2);
  f16*   n1wT  = (f16*)  carve((size_t)HID * 256 * 2);
  f16*   n2wT  = (f16*)  carve(HID * HID * 2);
  f16*   o1wT  = (f16*)  carve(HID * HID * 2);
  float* ohid  = agg;  // reuse agg buffer for readout hidden

  float* out = (float*)d_out;

  // ---- pipeline ----
  k_node_embed<<<(NN * HID + 255) / 256, 256, 0, stream>>>(typ, nembw, nembb, h, h16);
  k_rbf<<<(NE + 255) / 256, 256, 0, stream>>>(pos, esrc, edst, rbf);
  hipMemsetAsync(out, 0, NG * sizeof(float), stream);
  k_copy_pos<<<(NN * 3 + 255) / 256, 256, 0, stream>>>(pos, out);

  const int NPB = (MT_N + 7) / 8;  // 391 blocks of 8 waves for node-tile kernels
  for (int l = 0; l < 3; ++l) {
    k_prep_layer<<<128, 256, 0, stream>>>(lp[l].e1w, lp[l].e1b, lp[l].e2w, lp[l].n1w,
                                          lp[l].n2w, eemb, w1aT, w1bT, w1c, t4,
                                          e2wT, n1wT, n2wT);
    k_node_proj<<<NPB, 256, 0, stream>>>(h16, w1aT, w1bT, Pd, Ps);
    hipMemsetAsync(agg, 0, (size_t)NN * HID * 4, stream);
    k_edge<<<2000, 256, 0, stream>>>(Pd, Ps, rbf, esrc, edst, etyp, w1c, t4, e2wT,
                                     lp[l].e2b, agg);
    k_node_hidden<<<NPB, 256, 0, stream>>>(h16, agg, n1wT, lp[l].n1b, hid16);
    k_node_out<<<NPB, 256, 0, stream>>>(hid16, n2wT, lp[l].n2b, h, h16);
  }

  k_prep_out<<<(HID * HID + 255) / 256, 256, 0, stream>>>(o1w, o1wT);
  k_out_hidden<<<NPB, 256, 0, stream>>>(h16, o1wT, o1b, ohid);
  k_readout<<<(NN + 255) / 256, 256, 0, stream>>>(ohid, o2w, o2b, batch, out);
}
